// MDTA_35923106464230
// MI455X (gfx1250) — matrix operand order repaired and checked by reading, not compiled, not run on hardware
//
#include <hip/hip_runtime.h>
#include <math.h>

// ---------------- problem constants ----------------
#define BB     4
#define CC     384
#define HH     128
#define WWID   128
#define HWD    (HH * WWID)        // 16384
#define NHEADS 8
#define DDIM   48                 // CC / NHEADS
#define NPIX   (BB * HWD)         // 65536
#define EPSV   1e-5f

typedef __attribute__((ext_vector_type(16))) _Float16 v16h;
typedef __attribute__((ext_vector_type(8)))  _Float16 v8h;
typedef __attribute__((ext_vector_type(8)))  float    v8f;

// ---------------- WMMA fragment helpers (wave32, gfx1250 layouts) -----------
// A matrix 16x32 f16: lanes 0-15 M=0..15; halves 0..15 -> K=hi*16+0..15
// (hi = lane>>4), matching load_bt_frag. Source row-major (m, k):
// 16 contiguous halves per lane -> 2x global_load_b128.
__device__ __forceinline__ v16h load_a_frag(const _Float16* __restrict__ base, int stride) {
    int lane = threadIdx.x & 31;
    int r    = lane & 15;
    int hi   = lane >> 4;
    const _Float16* row = base + (size_t)r * stride;
    v16h a;
#pragma unroll
    for (int t = 0; t < 16; ++t) a[t] = row[hi * 16 + t];
    return a;
}

// B matrix 32x16 f16: lane n (0..15) = column n holds K=0..15, lanes 16..31 hold
// K=16..31. Source stored (n, k) row-major with row stride strideN ->
// 16 contiguous halves per lane (2x global_load_b128).
__device__ __forceinline__ v16h load_bt_frag(const _Float16* __restrict__ base, int strideN) {
    int lane = threadIdx.x & 31;
    int n    = lane & 15;
    int k0   = (lane >> 4) * 16;
    const _Float16* p = base + (size_t)n * strideN + k0;
    v16h b;
#pragma unroll
    for (int t = 0; t < 16; ++t) b[t] = p[t];
    return b;
}

// C/D f32 16x16: VGPR g holds row m = g + 8*hi, column n = lane&15.
__device__ __forceinline__ void store_c_raw(float* __restrict__ out, int stride, const v8f& c) {
    int lane = threadIdx.x & 31;
    int n    = lane & 15;
    int hi   = lane >> 4;
#pragma unroll
    for (int g = 0; g < 8; ++g) out[(size_t)(g + 8 * hi) * stride + n] = c[g];
}

// Store C tile into pixel-major f16 layout: per lane 8 contiguous channels -> b128.
__device__ __forceinline__ void store_cT_f16(_Float16* __restrict__ outPM, int c0, const v8f& c) {
    int lane = threadIdx.x & 31;
    int n    = lane & 15;
    int hi   = lane >> 4;
    _Float16* p = outPM + (size_t)n * CC + c0 + hi * 8;
#pragma unroll
    for (int g = 0; g < 8; ++g) p[g] = (_Float16)c[g];
}

__device__ __forceinline__ void store_cT_f16_bias(_Float16* __restrict__ outPM, int c0, const v8f& c,
                                                  const float* __restrict__ bias) {
    int lane = threadIdx.x & 31;
    int n    = lane & 15;
    int hi   = lane >> 4;
    _Float16*    p  = outPM + (size_t)n * CC + c0 + hi * 8;
    const float* bb = bias + c0 + hi * 8;
#pragma unroll
    for (int g = 0; g < 8; ++g) p[g] = (_Float16)(c[g] + bb[g]);
}

#define WMMA_F16(a, b, c) \
    __builtin_amdgcn_wmma_f32_16x16x32_f16(false, (a), false, (b), (short)0, (c), false, false)

// ---------------- kernel 1: per-pixel LayerNorm (writes pixel-major) ---------
__global__ __launch_bounds__(256) void ln_kernel(const float* __restrict__ x,
                                                 const float* __restrict__ gamma,
                                                 const float* __restrict__ beta,
                                                 float* __restrict__ xn32pm,
                                                 _Float16* __restrict__ xn16pm) {
    int p = blockIdx.x * blockDim.x + threadIdx.x;
    if (p >= NPIX) return;
    int b = p / HWD, hw = p % HWD;
    const float* xp = x + (size_t)b * CC * HWD + hw;   // strided read, lane-coalesced
    float s = 0.f, ss = 0.f;
    for (int c = 0; c < CC; ++c) {
        float v = xp[(size_t)c * HWD];
        s += v; ss += v * v;
    }
    float mu  = s * (1.f / CC);
    float var = ss * (1.f / CC) - mu * mu;
    float inv = rsqrtf(var + EPSV);
    float*    op = xn32pm + (size_t)p * CC;            // contiguous per-thread writes
    _Float16* hp = xn16pm + (size_t)p * CC;
    for (int c = 0; c < CC; ++c) {
        float v = (xp[(size_t)c * HWD] - mu) * inv * gamma[c] + beta[c];
        op[c] = v;
        hp[c] = (_Float16)v;
    }
}

// ---------------- kernel 2a: pack 4 pointwise weight matrices to f16 ---------
__global__ __launch_bounds__(256) void pack_w_kernel(const float* __restrict__ wq,
                                                     const float* __restrict__ wk,
                                                     const float* __restrict__ wv,
                                                     const float* __restrict__ wf,
                                                     _Float16* __restrict__ out) {
    int i = blockIdx.x * blockDim.x + threadIdx.x;
    if (i >= CC * CC) return;
    out[i]               = (_Float16)wq[i];
    out[CC * CC + i]     = (_Float16)wk[i];
    out[2 * CC * CC + i] = (_Float16)wv[i];
    out[3 * CC * CC + i] = (_Float16)wf[i];
}

// ---------------- kernel 2b: transpose dw weights to [mat][tap][C] -----------
__global__ __launch_bounds__(256) void pack_dw_kernel(const float* __restrict__ wq,
                                                      const float* __restrict__ wk,
                                                      const float* __restrict__ wv,
                                                      float* __restrict__ wdT) {
    int i = blockIdx.x * blockDim.x + threadIdx.x;
    if (i >= 9 * CC) return;
    int tap = i / CC, c = i % CC;
    wdT[i]              = wq[c * 9 + tap];
    wdT[9 * CC + i]     = wk[c * 9 + tap];
    wdT[2 * 9 * CC + i] = wv[c * 9 + tap];
}

// ---------------- kernel 3: fused QKV pointwise GEMM (WMMA) ------------------
// jobs = 3 mats * 24 o-tiles * 1024 pixel64-tiles = 73728 waves (exact /8)
// B operand pixel-major -> all fragment loads contiguous b128; output pixel-major.
__global__ __launch_bounds__(256) void qkv_gemm_kernel(const _Float16* __restrict__ xn16pm,
                                                       const _Float16* __restrict__ wpack,
                                                       const float* __restrict__ bq,
                                                       const float* __restrict__ bk,
                                                       const float* __restrict__ bv,
                                                       _Float16* __restrict__ qkv_pw) {
    int wid = blockIdx.x * (blockDim.x >> 5) + (threadIdx.x >> 5);
    int p64 = wid & 1023;
    int t   = wid >> 10;
    int ot  = t % 24;
    int mat = t / 24;
    if (mat >= 3) return;

    int p0 = p64 * 64;
    const _Float16* Xb    = xn16pm + (size_t)p0 * CC;
    const _Float16* Wbase = wpack + (size_t)mat * CC * CC + (size_t)ot * 16 * CC;
    const float*    bias  = (mat == 0) ? bq : (mat == 1 ? bk : bv);

    v8f acc0 = {}, acc1 = {}, acc2 = {}, acc3 = {};
#pragma unroll 2
    for (int kk = 0; kk < CC; kk += 32) {
        v16h a = load_a_frag(Wbase + kk, CC);
        acc0 = WMMA_F16(a, load_bt_frag(Xb + kk,                      CC), acc0);
        acc1 = WMMA_F16(a, load_bt_frag(Xb + (size_t)16 * CC + kk,    CC), acc1);
        acc2 = WMMA_F16(a, load_bt_frag(Xb + (size_t)32 * CC + kk,    CC), acc2);
        acc3 = WMMA_F16(a, load_bt_frag(Xb + (size_t)48 * CC + kk,    CC), acc3);
    }
    _Float16* out = qkv_pw + ((size_t)mat * NPIX + p0) * CC;
    store_cT_f16_bias(out,                      ot * 16, acc0, bias);
    store_cT_f16_bias(out + (size_t)16 * CC,    ot * 16, acc1, bias);
    store_cT_f16_bias(out + (size_t)32 * CC,    ot * 16, acc2, bias);
    store_cT_f16_bias(out + (size_t)48 * CC,    ot * 16, acc3, bias);
}

// ---------------- kernel 4: 3x3 depthwise conv -------------------------------
// Thread per (mat, pixel); pixel-major input, v8h vector loads along channels.
// Q,K written channel-major (for scores); V written pixel-major (for apply).
__global__ __launch_bounds__(256) void dw_kernel(const _Float16* __restrict__ in,   // [3][NPIX][CC]
                                                 const float* __restrict__ wdT,     // [3][9][CC]
                                                 const float* __restrict__ bq,
                                                 const float* __restrict__ bk,
                                                 const float* __restrict__ bv,
                                                 _Float16* __restrict__ outQcm,     // [BB][CC][HWD]
                                                 _Float16* __restrict__ outKcm,     // [BB][CC][HWD]
                                                 _Float16* __restrict__ outVpm) {   // [NPIX][CC]
    int tid = blockIdx.x * blockDim.x + threadIdx.x;
    int mat = tid / NPIX;                    // uniform per wave (NPIX % 256 == 0)
    int p   = tid % NPIX;
    if (mat >= 3) return;
    int b = p / HWD, hw = p % HWD;
    int h = hw / WWID, w = hw % WWID;
    const _Float16* base = in + ((size_t)mat * NPIX + p) * CC;
    const float*    wt   = wdT + mat * 9 * CC;
    const float*    bias = (mat == 0) ? bq : (mat == 1 ? bk : bv);

    bool valid[9]; int off[9];
#pragma unroll
    for (int dh = -1; dh <= 1; ++dh)
#pragma unroll
        for (int dw2 = -1; dw2 <= 1; ++dw2) {
            int t2 = (dh + 1) * 3 + (dw2 + 1);
            valid[t2] = (h + dh >= 0) && (h + dh < HH) && (w + dw2 >= 0) && (w + dw2 < WWID);
            off[t2]   = (dh * WWID + dw2) * CC;
        }

    for (int c0 = 0; c0 < CC; c0 += 8) {
        float acc[8];
#pragma unroll
        for (int j = 0; j < 8; ++j) acc[j] = bias[c0 + j];
#pragma unroll
        for (int t2 = 0; t2 < 9; ++t2) {
            if (!valid[t2]) continue;
            v8h val = *(const v8h*)(base + off[t2] + c0);       // 16B vector load
            const float* wrow = wt + t2 * CC + c0;              // contiguous weights
#pragma unroll
            for (int j = 0; j < 8; ++j) acc[j] += wrow[j] * (float)val[j];
        }
        if (mat == 2) {
            v8h o;
#pragma unroll
            for (int j = 0; j < 8; ++j) o[j] = (_Float16)acc[j];
            *(v8h*)(outVpm + (size_t)p * CC + c0) = o;          // vector store
        } else {
            _Float16* oc = (mat == 0 ? outQcm : outKcm) + (size_t)b * CC * HWD + hw;
#pragma unroll
            for (int j = 0; j < 8; ++j)                          // lane-coalesced scatter
                oc[(size_t)(c0 + j) * HWD] = (_Float16)acc[j];
        }
    }
}

// ---------------- kernel 5: attention scores S = K . Q^T (WMMA) --------------
// Q,K channel-major -> contraction dim (hw) contiguous for both operands.
// jobs = 8 k-chunks * 32 (b,head) * 3 * 3 tiles = 2304 waves (exact /8)
__global__ __launch_bounds__(256) void scores_kernel(const _Float16* __restrict__ Qcm,
                                                     const _Float16* __restrict__ Kcm,
                                                     float* __restrict__ partial) {
    int wid = blockIdx.x * (blockDim.x >> 5) + (threadIdx.x >> 5);
    int tj = wid % 3; int t = wid / 3;
    int ti = t % 3;   t /= 3;
    int bh = t & 31;  int chunk = t >> 5;
    if (chunk >= 8) return;
    int b = bh >> 3, head = bh & 7;
    const _Float16* Kb = Kcm + ((size_t)b * CC + head * DDIM + ti * 16) * HWD;
    const _Float16* Qb = Qcm + ((size_t)b * CC + head * DDIM + tj * 16) * HWD;

    int k0 = chunk * (HWD / 8);
    v8f acc = {};
    for (int kk = 0; kk < HWD / 8; kk += 32) {
        v16h a  = load_a_frag(Kb + k0 + kk, HWD);   // k contiguous along hw
        v16h bt = load_bt_frag(Qb + k0 + kk, HWD);  // contiguous per-lane loads
        acc = WMMA_F16(a, bt, acc);
    }
    float* out = partial + ((((size_t)chunk * 32 + bh) * 9) + ti * 3 + tj) * 256;
    store_c_raw(out, 16, acc);
}

// ---------------- kernel 6: deterministic chunk reduce + 1/alpha -------------
__global__ __launch_bounds__(256) void reduce_kernel(const float* __restrict__ partial,
                                                     const float* __restrict__ alpha,
                                                     float* __restrict__ scores) {
    int id = blockIdx.x * blockDim.x + threadIdx.x;
    if (id >= 32 * DDIM * DDIM) return;
    int j = id % DDIM; int t = id / DDIM;
    int i = t % DDIM;  int bh = t / DDIM;
    int ti = i >> 4, m = i & 15, tj = j >> 4, n = j & 15;
    float s = 0.f;
#pragma unroll
    for (int ch = 0; ch < 8; ++ch)
        s += partial[((((size_t)ch * 32 + bh) * 9) + ti * 3 + tj) * 256 + m * 16 + n];
    scores[((size_t)bh * DDIM + i) * DDIM + j] = s / alpha[bh & 7];
}

// ---------------- kernel 7: softmax rows -> transposed f16, K padded to 64 ---
__global__ __launch_bounds__(256) void softmax_kernel(const float* __restrict__ scores,
                                                      _Float16* __restrict__ softT) {
    int tid = blockIdx.x * blockDim.x + threadIdx.x;
    if (tid >= 32 * 64) return;
    int i = tid & 63, bh = tid >> 6;
    _Float16* dst = softT + (size_t)bh * DDIM * 64;
    if (i >= DDIM) {  // zero padding rows i = 48..63
        for (int j = 0; j < DDIM; ++j) dst[j * 64 + i] = (_Float16)0.f;
        return;
    }
    const float* row = scores + ((size_t)bh * DDIM + i) * DDIM;
    float mx = -3.4e38f;
    for (int j = 0; j < DDIM; ++j) mx = fmaxf(mx, row[j]);
    float s = 0.f;
    for (int j = 0; j < DDIM; ++j) s += __expf(row[j] - mx);
    float inv = 1.f / s;
    for (int j = 0; j < DDIM; ++j)
        dst[j * 64 + i] = (_Float16)(__expf(row[j] - mx) * inv);
}

// ---------------- kernel 8: O = soft^T . V (WMMA) ----------------------------
// V pixel-major -> B loads contiguous; O stored pixel-major (feeds final GEMM).
// jobs = 32 bh * 3 j-tiles * 256 hw64-tiles = 24576 waves (exact /8)
__global__ __launch_bounds__(256) void apply_kernel(const _Float16* __restrict__ Vpm,
                                                    const _Float16* __restrict__ softT,
                                                    _Float16* __restrict__ attnPM) {
    int wid  = blockIdx.x * (blockDim.x >> 5) + (threadIdx.x >> 5);
    int hw64 = wid & 255;
    int t    = wid >> 8;
    int tj   = t % 3;
    int bh   = t / 3;
    if (bh >= 32) return;
    int b = bh >> 3, head = bh & 7;
    int p0 = b * HWD + hw64 * 64;
    const _Float16* Ab = softT + (size_t)bh * DDIM * 64 + (size_t)tj * 16 * 64;
    const _Float16* Vb = Vpm + (size_t)p0 * CC + head * DDIM;

    v8f acc0 = {}, acc1 = {}, acc2 = {}, acc3 = {};
#pragma unroll
    for (int kk = 0; kk < 64; kk += 32) {
        v16h a = load_a_frag(Ab + kk, 64);
        const _Float16* Bb = Vb + kk;
        acc0 = WMMA_F16(a, load_bt_frag(Bb,                    CC), acc0);
        acc1 = WMMA_F16(a, load_bt_frag(Bb + (size_t)16 * CC,  CC), acc1);
        acc2 = WMMA_F16(a, load_bt_frag(Bb + (size_t)32 * CC,  CC), acc2);
        acc3 = WMMA_F16(a, load_bt_frag(Bb + (size_t)48 * CC,  CC), acc3);
    }
    _Float16* out = attnPM + (size_t)p0 * CC;
    int c0 = head * DDIM + tj * 16;
    store_cT_f16(out,                    c0, acc0);
    store_cT_f16(out + (size_t)16 * CC,  c0, acc1);
    store_cT_f16(out + (size_t)32 * CC,  c0, acc2);
    store_cT_f16(out + (size_t)48 * CC,  c0, acc3);
}

// ---------------- kernel 9: final 1x1 GEMM + bias + residual -----------------
// jobs = 24 o-tiles * 1024 pixel64-tiles = 24576 waves (exact /8)
__global__ __launch_bounds__(256) void final_gemm_kernel(const _Float16* __restrict__ attnPM,
                                                         const _Float16* __restrict__ wpack,
                                                         const float* __restrict__ bf,
                                                         const float* __restrict__ xn32pm,
                                                         float* __restrict__ out) {
    int wid = blockIdx.x * (blockDim.x >> 5) + (threadIdx.x >> 5);
    int p64 = wid & 1023;
    int ot  = wid >> 10;
    if (ot >= 24) return;
    int p0  = p64 * 64;
    int b   = p0 / HWD;
    int hw0 = p0 % HWD;
    const _Float16* Xb    = attnPM + (size_t)p0 * CC;
    const _Float16* Wbase = wpack + (size_t)3 * CC * CC + (size_t)ot * 16 * CC;

    v8f acc0 = {}, acc1 = {}, acc2 = {}, acc3 = {};
#pragma unroll 2
    for (int kk = 0; kk < CC; kk += 32) {
        v16h a = load_a_frag(Wbase + kk, CC);
        acc0 = WMMA_F16(a, load_bt_frag(Xb + kk,                   CC), acc0);
        acc1 = WMMA_F16(a, load_bt_frag(Xb + (size_t)16 * CC + kk, CC), acc1);
        acc2 = WMMA_F16(a, load_bt_frag(Xb + (size_t)32 * CC + kk, CC), acc2);
        acc3 = WMMA_F16(a, load_bt_frag(Xb + (size_t)48 * CC + kk, CC), acc3);
    }
    int lane = threadIdx.x & 31;
    int n    = lane & 15;
    int hi   = lane >> 4;
    v8f accs[4] = {acc0, acc1, acc2, acc3};
#pragma unroll
    for (int u = 0; u < 4; ++u) {
        const float* res = xn32pm + (size_t)(p0 + u * 16 + n) * CC + ot * 16 + hi * 8; // contiguous
        float* op = out + ((size_t)b * CC + ot * 16 + hi * 8) * HWD + hw0 + u * 16 + n;
#pragma unroll
        for (int g = 0; g < 8; ++g) {
            // row m = hi*8 + g, NCHW f32 output store (lane-coalesced, forced layout)
            op[(size_t)g * HWD] = accs[u][g] + bf[ot * 16 + hi * 8 + g] + res[g];
        }
    }
}

// ---------------- workspace layout (bytes) ----------------
static const size_t O_XN32   = 0;            // 100,663,296  xn f32 pixel-major
static const size_t O_XN16   = 100663296;    //  50,331,648  xn f16 pixel-major (reused as attn)
static const size_t O_WPACK  = 150994944;    //   1,179,648  4x pointwise weights f16
static const size_t O_WDT    = 152174592;    //      41,472  dw weights [3][9][C] f32
static const size_t O_QKVPW  = 152216064;    // 150,994,944  pointwise out, pixel-major
static const size_t O_QKVDW  = 303211008;    // 150,994,944  Qcm | Kcm | Vpm
static const size_t O_PART   = 454205952;    //   2,359,296  score partials
static const size_t O_SCORES = 456565248;    //     294,912
static const size_t O_SOFTT  = 456860160;    //     196,608
// total ~457,056,768 bytes

extern "C" void kernel_launch(void* const* d_in, const int* in_sizes, int n_in,
                              void* d_out, int out_size, void* d_ws, size_t ws_size,
                              hipStream_t stream) {
    const float* x        = (const float*)d_in[0];
    const float* ln_gamma = (const float*)d_in[1];
    const float* ln_beta  = (const float*)d_in[2];
    const float* wq_p = (const float*)d_in[3];
    const float* bq_p = (const float*)d_in[4];
    const float* wq_d = (const float*)d_in[5];
    const float* bq_d = (const float*)d_in[6];
    const float* wk_p = (const float*)d_in[7];
    const float* bk_p = (const float*)d_in[8];
    const float* wk_d = (const float*)d_in[9];
    const float* bk_d = (const float*)d_in[10];
    const float* wv_p = (const float*)d_in[11];
    const float* bv_p = (const float*)d_in[12];
    const float* wv_d = (const float*)d_in[13];
    const float* bv_d = (const float*)d_in[14];
    const float* alpha = (const float*)d_in[15];
    const float* wf   = (const float*)d_in[16];
    const float* bf   = (const float*)d_in[17];
    float* out = (float*)d_out;

    char* ws = (char*)d_ws;
    float*    xn32   = (float*)(ws + O_XN32);
    _Float16* xn16   = (_Float16*)(ws + O_XN16);
    _Float16* wpack  = (_Float16*)(ws + O_WPACK);
    float*    wdT    = (float*)(ws + O_WDT);
    _Float16* qkv_pw = (_Float16*)(ws + O_QKVPW);
    _Float16* qkv_dw = (_Float16*)(ws + O_QKVDW);
    float*    part   = (float*)(ws + O_PART);
    float*    scores = (float*)(ws + O_SCORES);
    _Float16* softT  = (_Float16*)(ws + O_SOFTT);
    _Float16* Qcm    = qkv_dw;
    _Float16* Kcm    = qkv_dw + (size_t)NPIX * CC;
    _Float16* Vpm    = qkv_dw + (size_t)2 * NPIX * CC;
    _Float16* attn   = xn16;  // xn16 dead after QKV GEMM; exact-size reuse

    // 1) LayerNorm -> xn32 (residual) + xn16 (GEMM operand), pixel-major
    ln_kernel<<<NPIX / 256, 256, 0, stream>>>(x, ln_gamma, ln_beta, xn32, xn16);

    // 2) weight packing: pointwise -> f16; depthwise -> [tap][C] f32
    pack_w_kernel<<<(CC * CC + 255) / 256, 256, 0, stream>>>(wq_p, wk_p, wv_p, wf, wpack);
    pack_dw_kernel<<<(9 * CC + 255) / 256, 256, 0, stream>>>(wq_d, wk_d, wv_d, wdT);

    // 3) fused QKV pointwise GEMM (WMMA f16), bias fused, pixel-major in/out
    qkv_gemm_kernel<<<73728 / 8, 256, 0, stream>>>(xn16, wpack, bq_p, bk_p, bv_p, qkv_pw);

    // 4) depthwise 3x3 conv; Q,K -> channel-major, V -> pixel-major
    dw_kernel<<<(3 * NPIX) / 256, 256, 0, stream>>>(
        qkv_pw, wdT, bq_d, bk_d, bv_d, Qcm, Kcm, Vpm);

    // 5) channel-attention scores: K . Q^T, K-dim split into 8 chunks (WMMA)
    scores_kernel<<<2304 / 8, 256, 0, stream>>>(Qcm, Kcm, part);

    // 6) deterministic chunk reduction + 1/alpha scaling
    reduce_kernel<<<(32 * DDIM * DDIM) / 256, 256, 0, stream>>>(part, alpha, scores);

    // 7) softmax rows -> transposed, K-padded f16 operand
    softmax_kernel<<<(32 * 64) / 256, 256, 0, stream>>>(scores, softT);

    // 8) O = soft^T . V (WMMA), pixel-major out
    apply_kernel<<<24576 / 8, 256, 0, stream>>>(Vpm, softT, attn);

    // 9) final 1x1 GEMM + bias + layernormed residual -> f32 NCHW output
    final_gemm_kernel<<<24576 / 8, 256, 0, stream>>>(attn, wpack, bf, xn32, out);
}
